// Gated_Atten_34832184770960
// MI455X (gfx1250) — matrix operand order repaired and checked by reading, not compiled, not run on hardware
//
#include <hip/hip_runtime.h>
#include <hip/hip_bf16.h>

typedef __attribute__((ext_vector_type(16))) _Float16 v16h;
typedef __attribute__((ext_vector_type(2)))  _Float16 v2h;
typedef __attribute__((ext_vector_type(8)))  float    v8f;

#define C_DIM 128
#define H_DIM 64
#define B_DIM 8
#define LX_DIM 128
#define LY_DIM 1024

// ---------------- fragment coordinate helpers (CDNA5 WMMA f16, wave32) --------
// A matrix 16x32 (MxK), 16-bit: lane holds row m = lane&15.
// Half pair vg (0..7) covers k = kbase(vg), kbase(vg)+1:
__device__ __forceinline__ int a_kbase(int lane, int vg) {
  return ((vg & 4) ? 16 : 0) + ((lane & 16) ? 8 : 0) + ((vg & 3) << 1);
}
// B matrix 32x16 (KxN), 16-bit: lane holds col n = lane&15; half j -> k:
__device__ __forceinline__ int b_k(int lane, int j) {
  return ((j & 8) ? 16 : 0) + ((lane & 16) ? 8 : 0) + (j & 7);
}
// C/D 16x16 f32: vgpr v -> m = v + ((lane&16)?8:0), n = lane&15.

// ---------------- kernel 0: zero accumulators --------------------------------
__global__ void zero_kernel(float* __restrict__ p, int n) {
  int i = blockIdx.x * blockDim.x + threadIdx.x;
  if (i < n) p[i] = 0.0f;
}

// ---------------- kernel 1: att projections via WMMA -------------------------
// out[b,l,c] = sum_k in[b,k,l] * W[k,c] + bias[c],  out stored f16 [B][L][C].
// One wave per 16-position tile: D[16 x 128] = A(X^T)[16 x 128] * B(W)[128 x 128].
__global__ void __launch_bounds__(32, 1)
att_wmma_kernel(const float* __restrict__ in,    // [B][C][L]
                const float* __restrict__ W,     // [C][C]
                const float* __restrict__ bias,  // [C]
                _Float16* __restrict__ out,      // [B][L][C]
                int L) {
  const int lane = threadIdx.x;
  const int l0   = blockIdx.x * 16;
  const int b    = blockIdx.y;
  const int m    = lane & 15;
  const int n    = lane & 15;

  // A fragments: X^T tile, element (m=l, k=c'), gathered (transposed) from [B][C][L]
  v16h af[4];
#pragma unroll
  for (int kc = 0; kc < 4; ++kc)
#pragma unroll
    for (int vg = 0; vg < 8; ++vg) {
      int k0 = kc * 32 + a_kbase(lane, vg);
      af[kc][2 * vg]     = (_Float16)in[((size_t)(b * C_DIM + k0)) * L + l0 + m];
      af[kc][2 * vg + 1] = (_Float16)in[((size_t)(b * C_DIM + k0 + 1)) * L + l0 + m];
    }

#pragma unroll
  for (int t = 0; t < 8; ++t) {
    float bv = bias[t * 16 + n];
    v8f acc;
#pragma unroll
    for (int i = 0; i < 8; ++i) acc[i] = bv;
#pragma unroll
    for (int kc = 0; kc < 4; ++kc) {
      v16h bf;  // W fragment: (k, n) -> W[k*128 + 16t + n], coalesced over lanes
#pragma unroll
      for (int j = 0; j < 16; ++j) {
        int k = kc * 32 + b_k(lane, j);
        bf[j] = (_Float16)W[k * C_DIM + t * 16 + n];
      }
      acc = __builtin_amdgcn_wmma_f32_16x16x32_f16(
          false, af[kc], false, bf, (short)0, acc, false, false);
    }
#pragma unroll
    for (int v = 0; v < 8; ++v) {
      int mm = v + ((lane & 16) ? 8 : 0);
      out[((size_t)(b * L + l0 + mm)) * C_DIM + t * 16 + n] = (_Float16)acc[v];
    }
  }
}

// ---------------- kernel 2: fused pairwise MLP + reductions ------------------
__global__ void __launch_bounds__(128, 1)
main_kernel(const _Float16* __restrict__ d_att,  // [B][LX][C]
            const _Float16* __restrict__ p_att,  // [B][LY][C]
            const float* __restrict__ W_down,    // [128][64]
            const float* __restrict__ b_down,    // [64]
            const float* __restrict__ W_up,      // [64][128]
            const float* __restrict__ b_up,      // [128]
            float* __restrict__ comp_acc,        // [B][LX][C]  (atomic)
            float* __restrict__ prot_acc)        // [B][LY][C]  (atomic)
{
  __shared__ _Float16 lds_h[4][16 * H_DIM];  // per-wave H staging (2KB each)

  const int lane = threadIdx.x & 31;
  const int wave = threadIdx.x >> 5;
  const int ly0  = blockIdx.x * 16;          // 64 ly tiles
  const int b    = blockIdx.y;               // 8 batches
  const int n    = lane & 15;                // B/C fragment column
  const int m    = lane & 15;                // A fragment row

  const v16h zero16 = 0;
  const v8f  zero8  = 0;

  // ---- persistent B fragments: W_down [kc=0..3][t=0..3]
  v16h wd[4][4];
#pragma unroll
  for (int kc = 0; kc < 4; ++kc)
#pragma unroll
    for (int t = 0; t < 4; ++t)
#pragma unroll
      for (int j = 0; j < 16; ++j) {
        int k = kc * 32 + b_k(lane, j);
        wd[kc][t][j] = (_Float16)W_down[k * H_DIM + t * 16 + n];
      }
  // ---- persistent B fragments: W_up [kc=0..1][t=0..7]
  v16h wu[2][8];
#pragma unroll
  for (int kc = 0; kc < 2; ++kc)
#pragma unroll
    for (int t = 0; t < 8; ++t)
#pragma unroll
      for (int j = 0; j < 16; ++j) {
        int k = kc * 32 + b_k(lane, j);
        wu[kc][t][j] = (_Float16)W_up[k * C_DIM + t * 16 + n];
      }
  // ---- prot-att A fragments for this ly tile (fixed per block), pairwise loads
  v16h pf[4];
  {
    const _Float16* prow = p_att + ((size_t)(b * LY_DIM + ly0 + m)) * C_DIM;
#pragma unroll
    for (int kc = 0; kc < 4; ++kc)
#pragma unroll
      for (int vg = 0; vg < 8; ++vg) {
        v2h p2 = *(const v2h*)(prow + kc * 32 + a_kbase(lane, vg));
        pf[kc][2 * vg]     = p2[0];
        pf[kc][2 * vg + 1] = p2[1];
      }
  }
  // ---- bias C fragments
  v8f cbd[4], cbu[8];
#pragma unroll
  for (int t = 0; t < 4; ++t) {
    float v = b_down[t * 16 + n];
#pragma unroll
    for (int i = 0; i < 8; ++i) cbd[t][i] = v;
  }
#pragma unroll
  for (int t = 0; t < 8; ++t) {
    float v = b_up[t * 16 + n];
#pragma unroll
    for (int i = 0; i < 8; ++i) cbu[t][i] = v;
  }

  // ---- prot accumulator (sum over this wave's lx values)
  v8f pacc[8];
#pragma unroll
  for (int t = 0; t < 8; ++t) pacc[t] = zero8;

  _Float16* hb = lds_h[wave];

  for (int lx = wave; lx < LX_DIM; lx += 4) {
    // ---- S = relu(d[lx] broadcast + P_tile), built directly in A layout
    const _Float16* drow = d_att + ((size_t)(b * LX_DIM + lx)) * C_DIM;
    v16h sf[4];
#pragma unroll
    for (int kc = 0; kc < 4; ++kc) {
      v16h dv;
#pragma unroll
      for (int vg = 0; vg < 8; ++vg) {
        v2h d2 = *(const v2h*)(drow + kc * 32 + a_kbase(lane, vg));
        dv[2 * vg]     = d2[0];
        dv[2 * vg + 1] = d2[1];
      }
      sf[kc] = __builtin_elementwise_max(dv + pf[kc], zero16);  // v_pk_add + v_pk_max
    }
    // ---- GEMM1: H[16x64] = S[16x128] * W_down[128x64] + b_down
    v8f hacc[4];
#pragma unroll
    for (int t = 0; t < 4; ++t) {
      v8f acc = cbd[t];
#pragma unroll
      for (int kc = 0; kc < 4; ++kc)
        acc = __builtin_amdgcn_wmma_f32_16x16x32_f16(
            false, sf[kc], false, wd[kc][t], (short)0, acc, false, false);
      hacc[t] = __builtin_elementwise_max(acc, zero8);          // relu (v_max_f32)
    }
    // ---- H -> LDS (f16), cross-lane transpose into A layout
#pragma unroll
    for (int t = 0; t < 4; ++t)
#pragma unroll
      for (int v = 0; v < 8; ++v) {
        int mm = v + ((lane & 16) ? 8 : 0);
        hb[mm * H_DIM + t * 16 + n] = (_Float16)hacc[t][v];
      }
    __builtin_amdgcn_wave_barrier();
    asm volatile("s_wait_dscnt 0" ::: "memory");   // wave-local LDS RAW
    v16h hf[2];
#pragma unroll
    for (int kc = 0; kc < 2; ++kc)
#pragma unroll
      for (int vg = 0; vg < 8; ++vg) {
        v2h h2 = *(const v2h*)(hb + m * H_DIM + kc * 32 + a_kbase(lane, vg));
        hf[kc][2 * vg]     = h2[0];
        hf[kc][2 * vg + 1] = h2[1];
      }
    __builtin_amdgcn_wave_barrier();
    // ---- GEMM2: A[16x128] = H[16x64] * W_up[64x128] + b_up ; reduce
#pragma unroll
    for (int t = 0; t < 8; ++t) {
      v8f acc = cbu[t];
      acc = __builtin_amdgcn_wmma_f32_16x16x32_f16(
          false, hf[0], false, wu[0][t], (short)0, acc, false, false);
      acc = __builtin_amdgcn_wmma_f32_16x16x32_f16(
          false, hf[1], false, wu[1][t], (short)0, acc, false, false);
      pacc[t] += acc;                                // prot: sum over lx (regs)
      float s = 0.0f;                                // comp: sum over ly tile
#pragma unroll
      for (int i = 0; i < 8; ++i) s += acc[i];
      s += __shfl_xor(s, 16, 32);                    // combine the two m-halves
      if (lane < 16)
        atomicAdd(&comp_acc[((size_t)(b * LX_DIM + lx)) * C_DIM + t * 16 + n], s);
    }
  }

  // ---- flush prot accumulator (4 waves contribute, 32 lx values each)
#pragma unroll
  for (int t = 0; t < 8; ++t)
#pragma unroll
    for (int v = 0; v < 8; ++v) {
      int mm = v + ((lane & 16) ? 8 : 0);
      atomicAdd(&prot_acc[((size_t)(b * LY_DIM + ly0 + mm)) * C_DIM + t * 16 + n],
                pacc[t][v]);
    }
}

// ---------------- kernel 3: gates + output -----------------------------------
__global__ void gate_kernel(const float* __restrict__ x, const float* __restrict__ y,
                            const float* __restrict__ comp_acc,
                            const float* __restrict__ prot_acc,
                            float* __restrict__ out) {
  const int N1 = B_DIM * C_DIM * LX_DIM;                 // 131072
  const int N2 = B_DIM * C_DIM * LY_DIM;                 // 1048576
  int idx = blockIdx.x * blockDim.x + threadIdx.x;
  if (idx < N1) {
    int l = idx & (LX_DIM - 1);
    int c = (idx >> 7) & (C_DIM - 1);
    int b = idx >> 14;
    float g = comp_acc[((size_t)(b * LX_DIM + l)) * C_DIM + c] * (1.0f / LY_DIM);
    float gate = (1.0f / (1.0f + __expf(-g))) * tanhf(g);
    out[idx] = x[idx] * (0.5f + gate);
  } else if (idx < N1 + N2) {
    int j = idx - N1;
    int l = j & (LY_DIM - 1);
    int c = (j >> 10) & (C_DIM - 1);
    int b = j >> 17;
    float g = prot_acc[((size_t)(b * LY_DIM + l)) * C_DIM + c] * (1.0f / LX_DIM);
    float gate = (1.0f / (1.0f + __expf(-g))) * tanhf(g);
    out[idx] = y[j] * (0.5f + gate);
  }
}

// ---------------- launch -----------------------------------------------------
extern "C" void kernel_launch(void* const* d_in, const int* in_sizes, int n_in,
                              void* d_out, int out_size, void* d_ws, size_t ws_size,
                              hipStream_t stream) {
  (void)in_sizes; (void)n_in; (void)out_size; (void)ws_size;
  const float* x      = (const float*)d_in[0];
  const float* y      = (const float*)d_in[1];
  const float* W_drug = (const float*)d_in[2];
  const float* b_drug = (const float*)d_in[3];
  const float* W_prot = (const float*)d_in[4];
  const float* b_prot = (const float*)d_in[5];
  const float* W_down = (const float*)d_in[6];
  const float* b_down = (const float*)d_in[7];
  const float* W_up   = (const float*)d_in[8];
  const float* b_up   = (const float*)d_in[9];

  // workspace layout (bytes):
  //   [0,       262144)  d_att f16  [8][128][128]
  //   [262144, 2359296)  p_att f16  [8][1024][128]
  //   [2359296,2883584)  comp_acc f32 [8][128][128]
  //   [2883584,7077888)  prot_acc f32 [8][1024][128]
  char* ws = (char*)d_ws;
  _Float16* d_att   = (_Float16*)(ws);
  _Float16* p_att   = (_Float16*)(ws + 262144);
  float*    comp_ac = (float*)(ws + 2359296);
  float*    prot_ac = (float*)(ws + 2883584);

  const int NACC = 131072 + 1048576;  // comp + prot contiguous
  zero_kernel<<<(NACC + 255) / 256, 256, 0, stream>>>(comp_ac, NACC);

  // projections as WMMA GEMMs: one wave per 16-position tile
  att_wmma_kernel<<<dim3(LX_DIM / 16, B_DIM), 32, 0, stream>>>(x, W_drug, b_drug,
                                                               d_att, LX_DIM);
  att_wmma_kernel<<<dim3(LY_DIM / 16, B_DIM), 32, 0, stream>>>(y, W_prot, b_prot,
                                                               p_att, LY_DIM);

  dim3 grid(LY_DIM / 16, B_DIM);      // 64 x 8 blocks, 4 waves each
  main_kernel<<<grid, 128, 0, stream>>>(d_att, p_att, W_down, b_down, W_up, b_up,
                                        comp_ac, prot_ac);

  const int NOUT = 131072 + 1048576;
  gate_kernel<<<(NOUT + 255) / 256, 256, 0, stream>>>(x, y, comp_ac, prot_ac,
                                                      (float*)d_out);
}
